// DeformableTransformerFusionDecoderLayer_45767171506678
// MI455X (gfx1250) — matrix operand order repaired and checked by reading, not compiled, not run on hardware
//
#include <hip/hip_runtime.h>

// Problem constants (from the reference)
constexpr int Bb = 8, Qq = 900, Ss = 16384, Dd = 256, DFf = 1024, NHh = 8, NPp = 4, DHh = 32;

#define USE_ASYNC_LDS 1

typedef __attribute__((ext_vector_type(16))) __bf16        v16bf;
typedef __attribute__((ext_vector_type(8)))  float         v8f;
typedef __attribute__((ext_vector_type(8)))  unsigned int  v8u;

__device__ __forceinline__ unsigned short f2bf(float f) {
  union { float f; unsigned u; } c; c.f = f;
  unsigned u = c.u;
  u += 0x7FFFu + ((u >> 16) & 1u);   // round-to-nearest-even
  return (unsigned short)(u >> 16);
}
__device__ __forceinline__ unsigned pack2bf(float lo, float hi) {
  return (unsigned)f2bf(lo) | ((unsigned)f2bf(hi) << 16);
}

// ---------------------------------------------------------------------------
// Weight pre-pass: Wt[n][k] = bf16(W[k][n]).
// ---------------------------------------------------------------------------
__global__ __launch_bounds__(256) void wt_transpose(
    const float* __restrict__ W, unsigned short* __restrict__ Wt, int K, int N)
{
  __shared__ float tile[32][33];
  int k0 = blockIdx.y * 32, n0 = blockIdx.x * 32;
  int tx = threadIdx.x & 31, ty = threadIdx.x >> 5;   // 32 x 8
  for (int r = ty; r < 32; r += 8)
    tile[r][tx] = W[(size_t)(k0 + r) * N + (n0 + tx)];
  __syncthreads();
  for (int r = ty; r < 32; r += 8)
    Wt[(size_t)(n0 + r) * K + (k0 + tx)] = f2bf(tile[tx][r]);
}

// ---------------------------------------------------------------------------
// Double-buffered bf16 WMMA GEMM: out[M,N] = epi(A[M,K] @ W[K,N] + bias),
// W pre-transposed/bf16 as Wt[N][K].
// Block: 256 threads = 8 waves, 128x128 C-tile, BK=64 (2 wmma K-steps/stage).
// Stage s+1's B slab goes through the async unit (global_load_async_to_lds)
// and A slab through float4 register loads while stage s computes 16 WMMAs.
// EPI: 0 = bias, 1 = bias+relu, 2 = out = add_in + sigmoid(v)*mul_in (gate)
// ---------------------------------------------------------------------------
template <int EPI>
__global__ __launch_bounds__(256) void gemm_bf16(
    const float* __restrict__ A, const unsigned short* __restrict__ Wt,
    const float* __restrict__ bias, float* __restrict__ out,
    int M, int N, int K,
    const float* __restrict__ add_in, const float* __restrict__ mul_in)
{
  constexpr int BM = 128, BN = 128, BK = 64;
  __shared__ __align__(16) unsigned short As[2][BM * BK];  // [m][k]
  __shared__ __align__(16) unsigned short Bs[2][BN * BK];  // [n][k]

  const int t    = threadIdx.x;
  const int lane = t & 31, wid = t >> 5;
  const int row0 = blockIdx.y * BM;
  const int col0 = blockIdx.x * BN;
  const int wm   = (wid & 3) * 32;
  const int wn   = (wid >> 2) * 64;
  const bool full = (row0 + BM) <= M;   // block-uniform: no predication path

  v8f acc[2][4];
#pragma unroll
  for (int mi = 0; mi < 2; mi++)
#pragma unroll
    for (int ni = 0; ni < 4; ni++)
#pragma unroll
      for (int r = 0; r < 8; r++) acc[mi][ni][r] = 0.f;

  // ---- staging helpers ----
  auto stageB = [&](int buf, int k0) {   // 128 x 64 bf16, 1024 16B chunks
#pragma unroll
    for (int i = 0; i < 4; i++) {
      int c = i * 256 + t;
      int n = c >> 3, k8 = (c & 7) * 8;
      const unsigned short* gp = Wt + (size_t)(col0 + n) * K + (k0 + k8);
#if USE_ASYNC_LDS
      asm volatile("global_load_async_to_lds_b128 %0, %1, off"
                   :: "v"((unsigned)(size_t)&Bs[buf][c * 8]),
                      "v"((unsigned long long)(size_t)gp)
                   : "memory");
#else
      *(uint4*)&Bs[buf][c * 8] = *(const uint4*)gp;
#endif
    }
  };
  auto loadA = [&](int k0, float4* r) {  // 128 x 64 fp32, 8 quads/thread
#pragma unroll
    for (int i = 0; i < 8; i++) {
      int c = i * 256 + t;
      int m = c >> 4, kq = (c & 15) * 4;
      if (full) {
        r[i] = *(const float4*)(A + (size_t)(row0 + m) * K + (k0 + kq));
      } else {
        int gr = row0 + m;
        int cr = gr < M ? gr : M - 1;    // clamped (always in-bounds) load
        r[i] = *(const float4*)(A + (size_t)cr * K + (k0 + kq));
        if (gr >= M) r[i] = make_float4(0.f, 0.f, 0.f, 0.f);
      }
    }
  };
  auto storeA = [&](int buf, const float4* r) {
#pragma unroll
    for (int i = 0; i < 8; i++) {
      int c = i * 256 + t;
      int m = c >> 4, kq = (c & 15) * 4;
      uint2 p; p.x = pack2bf(r[i].x, r[i].y); p.y = pack2bf(r[i].z, r[i].w);
      *(uint2*)&As[buf][m * BK + kq] = p;
    }
  };

  // ---- prologue: stage slab 0 ----
  float4 areg[8];
  stageB(0, 0);
  loadA(0, areg);
  storeA(0, areg);
#if USE_ASYNC_LDS
  asm volatile("s_wait_asynccnt 0" ::: "memory");
#endif
  __syncthreads();

  const int nstage = K / BK;
  for (int s = 0; s < nstage; s++) {
    const int cur = s & 1, nxt = cur ^ 1;
    const bool more = (s + 1) < nstage;
    if (more) {                      // overlap next slab with compute
      stageB(nxt, (s + 1) * BK);
      loadA((s + 1) * BK, areg);
    }
    // ---- compute: 2 wmma K-steps from current buffers ----
#pragma unroll
    for (int ks = 0; ks < 2; ks++) {
      const int kb = ks * 32;
      v16bf afrag[2];
#pragma unroll
      for (int mi = 0; mi < 2; mi++) {
        int row = wm + mi * 16 + (lane & 15);
        int klo = kb + (lane >> 4) * 8;
        uint4 lo = *(const uint4*)&As[cur][row * BK + klo];
        uint4 hi = *(const uint4*)&As[cur][row * BK + 16 + klo];
        v8u u = {lo.x, lo.y, lo.z, lo.w, hi.x, hi.y, hi.z, hi.w};
        afrag[mi] = __builtin_bit_cast(v16bf, u);
      }
      v16bf bfrag[4];
#pragma unroll
      for (int ni = 0; ni < 4; ni++) {
        int col = wn + ni * 16 + (lane & 15);
        int kk = kb + (lane >> 4) * 8;
        uint4 lo = *(const uint4*)&Bs[cur][col * BK + kk];
        uint4 hi = *(const uint4*)&Bs[cur][col * BK + kk + 16];
        v8u u = {lo.x, lo.y, lo.z, lo.w, hi.x, hi.y, hi.z, hi.w};
        bfrag[ni] = __builtin_bit_cast(v16bf, u);
      }
#pragma unroll
      for (int mi = 0; mi < 2; mi++)
#pragma unroll
        for (int ni = 0; ni < 4; ni++)
          acc[mi][ni] = __builtin_amdgcn_wmma_f32_16x16x32_bf16(
              false, afrag[mi], false, bfrag[ni], (short)0, acc[mi][ni], false, false);
    }
    if (more) {
      storeA(nxt, areg);
#if USE_ASYNC_LDS
      asm volatile("s_wait_asynccnt 0" ::: "memory");
#endif
    }
    __syncthreads();
  }

  // Epilogue: C/D layout => VGPR r holds M = (lane>>4)*8 + r, N = lane&15
#pragma unroll
  for (int mi = 0; mi < 2; mi++) {
#pragma unroll
    for (int ni = 0; ni < 4; ni++) {
      int n = col0 + wn + ni * 16 + (lane & 15);
      float bb = bias ? bias[n] : 0.f;
      int mbase = row0 + wm + mi * 16 + ((lane >> 4) << 3);
#pragma unroll
      for (int r = 0; r < 8; r++) {
        int m = mbase + r;
        if (full || m < M) {
          float v = acc[mi][ni][r] + bb;
          size_t o = (size_t)m * N + n;
          if constexpr (EPI == 1) v = fmaxf(v, 0.f);
          if constexpr (EPI == 2) {
            float s = 1.f / (1.f + __expf(-v));
            v = add_in[o] + s * mul_in[o];
          }
          out[o] = v;
        }
      }
    }
  }
}

// ---------------------------------------------------------------------------
// q = q_feat + q_pos ; iq = q_i_feat + q_pos
// ---------------------------------------------------------------------------
__global__ __launch_bounds__(256) void make_q(
    const float* __restrict__ q_pos, const float* __restrict__ q_feat,
    const float* __restrict__ q_i, float* __restrict__ q, float* __restrict__ iq, int n)
{
  int i = blockIdx.x * blockDim.x + threadIdx.x;
  if (i < n) {
    float p = q_pos[i];
    q[i]  = q_feat[i] + p;
    iq[i] = q_i[i] + p;
  }
}

// ---------------------------------------------------------------------------
// Per-query (one wave each): off = q@Woff+boff (64), attn = softmax_4((q+iq)@Wattn+battn)
// ---------------------------------------------------------------------------
__global__ __launch_bounds__(256) void proj_softmax(
    const float* __restrict__ q, const float* __restrict__ iq,
    const float* __restrict__ Woff, const float* __restrict__ boff,
    const float* __restrict__ Wattn, const float* __restrict__ battn,
    float* __restrict__ off, float* __restrict__ attnw, int M)
{
  __shared__ float sq[8][256];
  __shared__ float sa[8][256];
  int lane = threadIdx.x & 31, wid = threadIdx.x >> 5;
  int j = blockIdx.x * 8 + wid;
  bool valid = j < M;
  int jj = valid ? j : 0;
  for (int i = lane; i < 256; i += 32) {
    float a = q[(size_t)jj * 256 + i];
    float b = iq[(size_t)jj * 256 + i];
    sq[wid][i] = a;
    sa[wid][i] = a + b;
  }
  __syncthreads();

  float a0 = 0.f, a1 = 0.f, c = 0.f;
  for (int k = 0; k < 256; k++) {
    float qk = sq[wid][k];
    a0 += qk * Woff[k * 64 + lane];
    a1 += qk * Woff[k * 64 + lane + 32];
    c  += sa[wid][k] * Wattn[k * 32 + lane];
  }
  c += battn[lane];
  float m4 = c;
  m4 = fmaxf(m4, __shfl_xor(m4, 1, 32));
  m4 = fmaxf(m4, __shfl_xor(m4, 2, 32));
  float e = __expf(c - m4);
  float s4 = e;
  s4 += __shfl_xor(s4, 1, 32);
  s4 += __shfl_xor(s4, 2, 32);
  float p = e / s4;
  if (valid) {
    off[(size_t)j * 64 + lane]      = a0 + boff[lane];
    off[(size_t)j * 64 + lane + 32] = a1 + boff[lane + 32];
    attnw[(size_t)j * 32 + lane]    = p;
  }
}

// ---------------------------------------------------------------------------
// Bilinear deformable sampling: one wave per (b,q,h), lane = channel d (DH=32).
// ---------------------------------------------------------------------------
__global__ __launch_bounds__(256) void deform_sample(
    const float* __restrict__ value, const float* __restrict__ refp,
    const float* __restrict__ off, const float* __restrict__ attnw,
    const int* __restrict__ shapes, float* __restrict__ out, int BQ)
{
  int lane = threadIdx.x & 31, wid = threadIdx.x >> 5;
  int g = blockIdx.x * 8 + wid;
  if (g >= BQ * NHh) return;
  int h  = g & (NHh - 1);
  int bq = g >> 3;
  int b  = bq / Qq;
  int Hi = shapes[0], Wi = shapes[1];
  float Hf = (float)Hi, Wf = (float)Wi;
  float rx = refp[(size_t)bq * 2 + 0];
  float ry = refp[(size_t)bq * 2 + 1];
  const float* vb = value + ((size_t)b * Ss) * Dd + h * DHh + lane;

  auto corner = [&](int xi, int yi, float cw) -> float {
    bool v = (xi >= 0) && (xi < Wi) && (yi >= 0) && (yi < Hi);
    int xc = xi < 0 ? 0 : (xi > Wi - 1 ? Wi - 1 : xi);
    int yc = yi < 0 ? 0 : (yi > Hi - 1 ? Hi - 1 : yi);
    float gv = vb[(size_t)(yc * Wi + xc) * Dd];
    return v ? cw * gv : 0.f;
  };

  float acc = 0.f;
#pragma unroll
  for (int p = 0; p < NPp; p++) {
    int oc = (h * NPp + p) * 2;
    float ox = off[(size_t)bq * 64 + oc];
    float oy = off[(size_t)bq * 64 + oc + 1];
    float w  = attnw[(size_t)bq * 32 + h * NPp + p];
    float x = (rx + ox / Wf) * Wf - 0.5f;
    float y = (ry + oy / Hf) * Hf - 0.5f;
    float x0f = floorf(x), y0f = floorf(y);
    float lx1 = x - x0f, ly1 = y - y0f;
    float lx0 = 1.f - lx1, ly0 = 1.f - ly1;
    int x0 = (int)x0f, y0 = (int)y0f;
    int x1 = x0 + 1, y1 = y0 + 1;
    acc += w * (corner(x0, y0, lx0 * ly0) + corner(x1, y0, lx1 * ly0) +
                corner(x0, y1, lx0 * ly1) + corner(x1, y1, lx1 * ly1));
  }
  out[(size_t)bq * Dd + h * DHh + lane] = acc;
}

// ---------------------------------------------------------------------------
// out = LayerNorm(x + y) * g + b   (one wave per 256-wide row)
// ---------------------------------------------------------------------------
__global__ __launch_bounds__(256) void ln_add(
    const float* __restrict__ x, const float* __restrict__ y,
    const float* __restrict__ g, const float* __restrict__ be,
    float* __restrict__ out, int M)
{
  int lane = threadIdx.x & 31, wid = threadIdx.x >> 5;
  int j = blockIdx.x * 8 + wid;
  if (j >= M) return;
  float v[8];
  float s = 0.f;
#pragma unroll
  for (int i = 0; i < 8; i++) {
    int c = lane + i * 32;
    float t = x[(size_t)j * 256 + c] + y[(size_t)j * 256 + c];
    v[i] = t; s += t;
  }
#pragma unroll
  for (int d = 16; d >= 1; d >>= 1) s += __shfl_xor(s, d, 32);
  float mean = s * (1.f / 256.f);
  float ss = 0.f;
#pragma unroll
  for (int i = 0; i < 8; i++) { float d = v[i] - mean; ss += d * d; }
#pragma unroll
  for (int d = 16; d >= 1; d >>= 1) ss += __shfl_xor(ss, d, 32);
  float rstd = rsqrtf(ss * (1.f / 256.f) + 1e-5f);
#pragma unroll
  for (int i = 0; i < 8; i++) {
    int c = lane + i * 32;
    out[(size_t)j * 256 + c] = (v[i] - mean) * rstd * g[c] + be[c];
  }
}

// ---------------------------------------------------------------------------
extern "C" void kernel_launch(void* const* d_in, const int* in_sizes, int n_in,
                              void* d_out, int out_size, void* d_ws, size_t ws_size,
                              hipStream_t stream)
{
  const float* src    = (const float*)d_in[0];
  const float* refp   = (const float*)d_in[1];
  const float* q_pos  = (const float*)d_in[2];
  const float* q_feat = (const float*)d_in[3];
  const float* q_i    = (const float*)d_in[4];
  const int*   shapes = (const int*)d_in[5];
  const float* Wv   = (const float*)d_in[7];  const float* bv   = (const float*)d_in[8];
  const float* Woff = (const float*)d_in[9];  const float* boff = (const float*)d_in[10];
  const float* Wat  = (const float*)d_in[11]; const float* bat  = (const float*)d_in[12];
  const float* Wout = (const float*)d_in[13]; const float* bout = (const float*)d_in[14];
  const float* n1g  = (const float*)d_in[15]; const float* n1b  = (const float*)d_in[16];
  const float* fg1w = (const float*)d_in[17]; const float* fg1b = (const float*)d_in[18];
  const float* fg2w = (const float*)d_in[19]; const float* fg2b = (const float*)d_in[20];
  const float* l1w  = (const float*)d_in[21]; const float* l1b  = (const float*)d_in[22];
  const float* l2w  = (const float*)d_in[23]; const float* l2b  = (const float*)d_in[24];
  const float* n2g  = (const float*)d_in[25]; const float* n2b  = (const float*)d_in[26];
  const float* l3w  = (const float*)d_in[27]; const float* l3b  = (const float*)d_in[28];
  const float* l4w  = (const float*)d_in[29]; const float* l4b  = (const float*)d_in[30];
  const float* n3g  = (const float*)d_in[31]; const float* n3b  = (const float*)d_in[32];

  const size_t BQ = (size_t)Bb * Qq;          // 7200
  float* ws    = (float*)d_ws;
  float* value = ws;                          // B*S*D floats (reused as FFN scratch later)
  float* q     = value + (size_t)Bb * Ss * Dd;
  float* iq    = q   + BQ * Dd;
  float* offb  = iq  + BQ * Dd;
  float* attb  = offb + BQ * 64;
  float* ao    = attb + BQ * 32;
  float* s2    = ao  + BQ * Dd;
  float* qi    = s2  + BQ * Dd;
  float* qf    = qi  + BQ * Dd;
  float* qi2   = qf  + BQ * Dd;
  // bf16 transposed weights after the fp32 regions
  unsigned short* wvt = (unsigned short*)(qi2 + BQ * Dd);
  unsigned short* wot = wvt + 256 * 256;
  unsigned short* f1t = wot + 256 * 256;
  unsigned short* f2t = f1t + 256 * 256;
  unsigned short* l1t = f2t + 256 * 256;      // [1024][256]
  unsigned short* l2t = l1t + 1024 * 256;     // [256][1024]
  unsigned short* l3t = l2t + 1024 * 256;
  unsigned short* l4t = l3t + 1024 * 256;
  float* hidden = value;                      // reuse: value dead after sampling
  float* tmp    = value + BQ * DFf;

  float* out_qf = (float*)d_out;
  float* out_qi = out_qf + BQ * Dd;

  const int Mq = (int)BQ;
  const dim3 blk(256);

  // 0) weight bf16 transposes (Wt[n][k])
  wt_transpose<<<dim3(8, 8),  blk, 0, stream>>>(Wv,   wvt, 256, 256);
  wt_transpose<<<dim3(8, 8),  blk, 0, stream>>>(Wout, wot, 256, 256);
  wt_transpose<<<dim3(8, 8),  blk, 0, stream>>>(fg1w, f1t, 256, 256);
  wt_transpose<<<dim3(8, 8),  blk, 0, stream>>>(fg2w, f2t, 256, 256);
  wt_transpose<<<dim3(32, 8), blk, 0, stream>>>(l1w,  l1t, 256, 1024);
  wt_transpose<<<dim3(8, 32), blk, 0, stream>>>(l2w,  l2t, 1024, 256);
  wt_transpose<<<dim3(32, 8), blk, 0, stream>>>(l3w,  l3t, 256, 1024);
  wt_transpose<<<dim3(8, 32), blk, 0, stream>>>(l4w,  l4t, 1024, 256);

  // 1) q = q_feat + q_pos, iq = q_i_feat + q_pos
  { int n = (int)(BQ * Dd); make_q<<<(n + 255) / 256, blk, 0, stream>>>(q_pos, q_feat, q_i, q, iq, n); }
  // 2) value = src @ Wv + bv   (M=131072, K=N=256)
  { dim3 g(Dd / 128, (Bb * Ss) / 128);
    gemm_bf16<0><<<g, blk, 0, stream>>>(src, wvt, bv, value, Bb * Ss, Dd, Dd, nullptr, nullptr); }
  // 3) offsets + softmaxed attention weights
  proj_softmax<<<(Mq + 7) / 8, blk, 0, stream>>>(q, iq, Woff, boff, Wat, bat, offb, attb, Mq);
  // 4) deformable bilinear sampling -> ao (B*Q, 256)
  deform_sample<<<((int)(BQ * NHh) + 7) / 8, blk, 0, stream>>>(value, refp, offb, attb, shapes, ao, Mq);
  // 5) src2 = ao @ Wout + bout
  { dim3 g(Dd / 128, (Mq + 127) / 128);
    gemm_bf16<0><<<g, blk, 0, stream>>>(ao, wot, bout, s2, Mq, Dd, Dd, nullptr, nullptr); }
  // 6) qi = LN(q_i_feat + src2)
  ln_add<<<(Mq + 7) / 8, blk, 0, stream>>>(q_i, s2, n1g, n1b, qi, Mq);
  // 7) qf = q_feat + sigmoid(qi@fg1+b)*qi
  { dim3 g(Dd / 128, (Mq + 127) / 128);
    gemm_bf16<2><<<g, blk, 0, stream>>>(qi, f1t, fg1b, qf, Mq, Dd, Dd, q_feat, qi); }
  // 8) qi2 = qi + sigmoid(q_feat@fg2+b)*q_feat
  { dim3 g(Dd / 128, (Mq + 127) / 128);
    gemm_bf16<2><<<g, blk, 0, stream>>>(q_feat, f2t, fg2b, qi2, Mq, Dd, Dd, qi, q_feat); }
  // 9) qi branch FFN + LN -> out_qi
  { dim3 g(DFf / 128, (Mq + 127) / 128);
    gemm_bf16<1><<<g, blk, 0, stream>>>(qi2, l1t, l1b, hidden, Mq, DFf, Dd, nullptr, nullptr); }
  { dim3 g(Dd / 128, (Mq + 127) / 128);
    gemm_bf16<0><<<g, blk, 0, stream>>>(hidden, l2t, l2b, tmp, Mq, Dd, DFf, nullptr, nullptr); }
  ln_add<<<(Mq + 7) / 8, blk, 0, stream>>>(qi2, tmp, n2g, n2b, out_qi, Mq);
  // 10) qf branch FFN + LN -> out_qf
  { dim3 g(DFf / 128, (Mq + 127) / 128);
    gemm_bf16<1><<<g, blk, 0, stream>>>(qf, l3t, l3b, hidden, Mq, DFf, Dd, nullptr, nullptr); }
  { dim3 g(Dd / 128, (Mq + 127) / 128);
    gemm_bf16<0><<<g, blk, 0, stream>>>(hidden, l4t, l4b, tmp, Mq, Dd, DFf, nullptr, nullptr); }
  ln_add<<<(Mq + 7) / 8, blk, 0, stream>>>(qf, tmp, n3g, n3b, out_qf, Mq);
}
